// HGT_59742995087910
// MI455X (gfx1250) — compile-verified, matrix operand order checked by reading, not run on hardware
//
#include <hip/hip_runtime.h>
#include <math.h>

// ---------------------------------------------------------------------------
// Types for CDNA5 WMMA (wave32, 16x16x32 bf16 -> f32) and TDM descriptors
// ---------------------------------------------------------------------------
typedef __attribute__((ext_vector_type(16))) __bf16   v16bf;
typedef __attribute__((ext_vector_type(8)))  float    v8f;
typedef __attribute__((ext_vector_type(4)))  unsigned v4u;
typedef __attribute__((ext_vector_type(4)))  int      v4i;
typedef __attribute__((ext_vector_type(8)))  int      v8i;

union BF16x16 {
    v16bf          v;
    unsigned short u[16];
    v4u            u4[2];
};

#if __has_builtin(__builtin_amdgcn_tensor_load_to_lds) && \
    __has_builtin(__builtin_amdgcn_s_wait_tensorcnt)
#define HAVE_TDM 1
#else
#define HAVE_TDM 0
#endif

__device__ __forceinline__ unsigned short f2bf(float f) {
    unsigned u = __float_as_uint(f);
    u += 0x7FFFu + ((u >> 16) & 1u);   // round-to-nearest-even
    return (unsigned short)(u >> 16);
}

__device__ __forceinline__ v8f zero8() {
    v8f z;
    #pragma unroll
    for (int i = 0; i < 8; ++i) z[i] = 0.0f;
    return z;
}

__device__ __forceinline__ v4u zero4u() {
    v4u z;
    #pragma unroll
    for (int i = 0; i < 4; ++i) z[i] = 0u;
    return z;
}

// ---------------------------------------------------------------------------
// Generic fill / fp32 -> bf16 convert
// ---------------------------------------------------------------------------
__global__ void fill_u32(unsigned* __restrict__ p, unsigned v, size_t n) {
    size_t i = (size_t)blockIdx.x * blockDim.x + threadIdx.x;
    if (i < n) p[i] = v;
}

__global__ void cvt_bf16(const float* __restrict__ in, unsigned short* __restrict__ out,
                         size_t n) {
    size_t i = (size_t)blockIdx.x * blockDim.x + threadIdx.x;
    if (i < n) out[i] = f2bf(in[i]);
}

// ---------------------------------------------------------------------------
// Prep: fuse relation transforms into projection weights, transpose, -> bf16
//   WqT[t][c][i]  = Wq[t][i][c]
//   WkpT[t][c][i] = sum_d Wk[t][i][h*16+d] * a_rel[t][h][d][e]   (c = h*16+e)
//   WvpT analogous with m_rel; fused biases in fp32.
// ---------------------------------------------------------------------------
__global__ void prep_qkv(const float* __restrict__ Wk, const float* __restrict__ bk,
                         const float* __restrict__ Wq, const float* __restrict__ bq,
                         const float* __restrict__ Wv, const float* __restrict__ bv,
                         const float* __restrict__ a_rel, const float* __restrict__ m_rel,
                         int din,
                         unsigned short* __restrict__ WqT,
                         unsigned short* __restrict__ WkpT,
                         unsigned short* __restrict__ WvpT,
                         float* __restrict__ bkp, float* __restrict__ bvp) {
    const int t = blockIdx.x;
    const float* wk = Wk + (size_t)t * din * 64;
    const float* wq = Wq + (size_t)t * din * 64;
    const float* wv = Wv + (size_t)t * din * 64;
    const float* ar = a_rel + (size_t)t * 1024;   // [4][16][16]
    const float* mr = m_rel + (size_t)t * 1024;
    unsigned short* oq = WqT  + (size_t)t * 64 * din;
    unsigned short* ok = WkpT + (size_t)t * 64 * din;
    unsigned short* ov = WvpT + (size_t)t * 64 * din;

    const int total = 64 * din;
    for (int idx = threadIdx.x; idx < total; idx += blockDim.x) {
        int c = idx / din, i = idx % din;
        int h = c >> 4, e = c & 15;
        oq[idx] = f2bf(wq[(size_t)i * 64 + c]);
        float sk = 0.f, sv = 0.f;
        #pragma unroll
        for (int d = 0; d < 16; ++d) {
            float a = ar[h * 256 + d * 16 + e];
            float m = mr[h * 256 + d * 16 + e];
            sk += wk[(size_t)i * 64 + h * 16 + d] * a;
            sv += wv[(size_t)i * 64 + h * 16 + d] * m;
        }
        ok[idx] = f2bf(sk);
        ov[idx] = f2bf(sv);
    }
    if (threadIdx.x < 64) {
        int c = threadIdx.x, h = c >> 4, e = c & 15;
        const float* bkt = bk + (size_t)t * 64;
        const float* bvt = bv + (size_t)t * 64;
        float sk = 0.f, sv = 0.f;
        #pragma unroll
        for (int d = 0; d < 16; ++d) {
            sk += bkt[h * 16 + d] * ar[h * 256 + d * 16 + e];
            sv += bvt[h * 16 + d] * mr[h * 256 + d * 16 + e];
        }
        bkp[t * 64 + c] = sk;
        bvp[t * 64 + c] = sv;
    }
    (void)bq;
}

__global__ void prep_wa(const float* __restrict__ Wa, unsigned short* __restrict__ WaT) {
    const int t = blockIdx.x;
    const float* w = Wa + (size_t)t * 64 * 64;
    unsigned short* o = WaT + (size_t)t * 64 * 64;
    for (int idx = threadIdx.x; idx < 4096; idx += blockDim.x) {
        int c = idx >> 6, i = idx & 63;
        o[idx] = f2bf(w[(size_t)i * 64 + c]);   // transpose
    }
}

// ---------------------------------------------------------------------------
// WMMA GEMM:  C[N,64] = post( A_bf16[N,K] @ B + bias )
//   Bt : bf16, pre-transposed [64][K] (column-major B), DMA-staged in LDS
//   via the Tensor Data Mover (TENSORcnt) when available.
//   Each wave computes a 16x64 tile with 4 accumulators; block = 4 waves.
//   mode: 0 = none, 1 = relu, 2 = sigmoid-skip blend with xin
// ---------------------------------------------------------------------------
__global__ __launch_bounds__(128) void gemm_wmma(
        const unsigned short* __restrict__ A, const unsigned short* __restrict__ Bt,
        const float* __restrict__ bias, float* __restrict__ C,
        int N, int K, int mode,
        const float* __restrict__ skip, const float* __restrict__ xin) {
    __shared__ unsigned short ldsB[64 * 128];

#if HAVE_TDM
    if (threadIdx.x < 32) {            // wave 0 issues one TDM descriptor
        const unsigned long long ga = (unsigned long long)Bt;
        const unsigned lds_off = (unsigned)(unsigned long long)(void*)&ldsB[0];
        const int elems = 64 * K;      // bf16 elements, flat 1-row tile

        v4u g0;                         // D# group 0
        g0[0] = 1u;                                                  // count=1 (user)
        g0[1] = lds_off;                                             // lds_addr
        g0[2] = (unsigned)(ga & 0xFFFFFFFFu);                        // global_addr[31:0]
        g0[3] = (unsigned)((ga >> 32) & 0x01FFFFFFu) | (2u << 30);   // addr[56:32], type=2

        v8i g1;                         // D# group 1
        #pragma unroll
        for (int i = 0; i < 8; ++i) g1[i] = 0;
        g1[0] = (1 << 16);                            // data_size=1 (2 bytes)
        g1[1] = (elems & 0xFFFF) << 16;               // tensor_dim0[15:0]
        g1[2] = ((unsigned)elems >> 16) | (1 << 16);  // tensor_dim0[31:16], tensor_dim1=1
        g1[3] = (elems & 0xFFFF) << 16;               // tile_dim0 = elems
        g1[5] = elems;                                // tensor_dim0_stride

        v4i gz;
        #pragma unroll
        for (int i = 0; i < 4; ++i) gz[i] = 0;

#if defined(__clang_major__) && (__clang_major__ >= 23)
        v8i gz8;
        #pragma unroll
        for (int i = 0; i < 8; ++i) gz8[i] = 0;
        __builtin_amdgcn_tensor_load_to_lds(g0, g1, gz, gz, gz8, 0);
#else
        __builtin_amdgcn_tensor_load_to_lds(g0, g1, gz, gz, 0);
#endif
        __builtin_amdgcn_s_wait_tensorcnt(0);
    }
#else
    {   // fallback: cooperative VMEM->VGPR->LDS stage of weights (64*K bf16)
        const unsigned* s = (const unsigned*)Bt;
        unsigned* d = (unsigned*)ldsB;
        const int total = (64 * K) >> 1;
        for (int i = threadIdx.x; i < total; i += blockDim.x) d[i] = s[i];
    }
#endif
    __syncthreads();

    const int wave    = threadIdx.x >> 5;
    const int lane    = threadIdx.x & 31;
    const int r       = lane & 15;
    const int halfsel = lane >> 4;       // 0: K 0-7/16-23, 1: K 8-15/24-31
    const int koff    = halfsel * 8;
    const int rowbase = blockIdx.x * 64 + wave * 16;

    v8f acc[4];
    #pragma unroll
    for (int cb = 0; cb < 4; ++cb) acc[cb] = zero8();

    const int  row   = rowbase + r;
    const bool rowok = row < N;
    const unsigned short* arow = A + (size_t)(rowok ? row : 0) * K;

    for (int k = 0; k < K; k += 32) {
        BF16x16 a;
        if (rowok) {
            a.u4[0] = *(const v4u*)(arow + k + koff);        // K k+koff .. +8
            a.u4[1] = *(const v4u*)(arow + k + 16 + koff);   // K k+16+koff .. +8
        } else {
            a.u4[0] = zero4u();
            a.u4[1] = zero4u();
        }
        #pragma unroll
        for (int cb = 0; cb < 4; ++cb) {
            const int col = cb * 16 + r;
            BF16x16 b;
            b.u4[0] = *(const v4u*)(ldsB + col * K + k + koff);
            b.u4[1] = *(const v4u*)(ldsB + col * K + k + 16 + koff);
            acc[cb] = __builtin_amdgcn_wmma_f32_16x16x32_bf16(
                false, a.v, false, b.v, (short)0, acc[cb], false, false);
        }
    }

    float skv = 0.f;
    if (mode == 2) skv = 1.f / (1.f + __expf(-skip[0]));

    #pragma unroll
    for (int cb = 0; cb < 4; ++cb) {
        const int col  = cb * 16 + r;
        const float bc = bias[col];
        #pragma unroll
        for (int j = 0; j < 8; ++j) {
            const int orow = rowbase + j + halfsel * 8;
            if (orow < N) {
                float v = acc[cb][j] + bc;
                if (mode == 1)      v = fmaxf(v, 0.f);
                else if (mode == 2) v = skv * v + (1.f - skv) * xin[(size_t)orow * 64 + col];
                C[(size_t)orow * 64 + col] = v;
            }
        }
    }
}

// ---------------------------------------------------------------------------
// Edge pass 1: alpha = <Q[dst,h], Kp[src,h]> * p_rel[h] / 4 ; atomicMax amax
// (float max via order-preserving uint encoding)
// ---------------------------------------------------------------------------
__global__ void edge_alpha(const float* __restrict__ Q, const float* __restrict__ Kp,
                           const int* __restrict__ src, const int* __restrict__ dst,
                           int E, const float* __restrict__ p_rel,
                           float* __restrict__ alpha, unsigned* __restrict__ amax) {
    int idx = blockIdx.x * blockDim.x + threadIdx.x;
    if (idx >= E * 4) return;
    const int e = idx >> 2, h = idx & 3;
    const int s = src[e], d = dst[e];
    const float4* q = (const float4*)(Q  + (size_t)d * 64 + h * 16);
    const float4* k = (const float4*)(Kp + (size_t)s * 64 + h * 16);
    float acc = 0.f;
    #pragma unroll
    for (int j = 0; j < 4; ++j) {
        float4 a = q[j], b = k[j];
        acc += a.x * b.x + a.y * b.y + a.z * b.z + a.w * b.w;
    }
    const float al = acc * p_rel[h] * 0.25f;
    alpha[idx] = al;
    const unsigned ub   = __float_as_uint(al);
    const unsigned skey = (ub & 0x80000000u) ? ~ub : (ub | 0x80000000u);
    atomicMax(&amax[(size_t)d * 4 + h], skey);
}

// ---------------------------------------------------------------------------
// Edge pass 2: w = exp(alpha - amax); den += w; agg += w * Vp[src]
// ---------------------------------------------------------------------------
__global__ void edge_accum(const float* __restrict__ Vp,
                           const int* __restrict__ src, const int* __restrict__ dst,
                           int E, const float* __restrict__ alpha,
                           const unsigned* __restrict__ amax,
                           float* __restrict__ den, float* __restrict__ agg) {
    int idx = blockIdx.x * blockDim.x + threadIdx.x;
    if (idx >= E * 4) return;
    const int e = idx >> 2, h = idx & 3;
    const int s = src[e], d = dst[e];
    const unsigned skey = amax[(size_t)d * 4 + h];
    const unsigned ub   = (skey & 0x80000000u) ? (skey & 0x7FFFFFFFu) : ~skey;
    float mx = __uint_as_float(ub);
    if (!isfinite(mx)) mx = 0.f;
    const float w = __expf(alpha[idx] - mx);
    atomicAdd(&den[(size_t)d * 4 + h], w);
    const float4* v = (const float4*)(Vp + (size_t)s * 64 + h * 16);
    float* o = agg + (size_t)d * 64 + h * 16;
    #pragma unroll
    for (int j = 0; j < 4; ++j) {
        float4 t = v[j];
        atomicAdd(o + j * 4 + 0, w * t.x);
        atomicAdd(o + j * 4 + 1, w * t.y);
        atomicAdd(o + j * 4 + 2, w * t.z);
        atomicAdd(o + j * 4 + 3, w * t.w);
    }
}

// ---------------------------------------------------------------------------
// Node finish: actb = bf16( gelu_exact(agg / (den + 1e-16)) )
// ---------------------------------------------------------------------------
__global__ void node_finish(const float* __restrict__ agg, const float* __restrict__ den,
                            unsigned short* __restrict__ actb, int N) {
    int idx = blockIdx.x * blockDim.x + threadIdx.x;
    if (idx >= N * 64) return;
    const int n = idx >> 6, h = (idx >> 4) & 3;
    const float x = agg[idx] / (den[(size_t)n * 4 + h] + 1e-16f);
    actb[idx] = f2bf(0.5f * x * (1.f + erff(x * 0.70710678118654752f)));
}

// ---------------------------------------------------------------------------
// Classifier: out[n] = h[n,:] @ Wc + bc
// ---------------------------------------------------------------------------
__global__ void classify(const float* __restrict__ h, const float* __restrict__ Wc,
                         const float* __restrict__ bc, float* __restrict__ out, int N) {
    int n = blockIdx.x * blockDim.x + threadIdx.x;
    if (n >= N) return;
    float acc = bc[0];
    const float* r = h + (size_t)n * 64;
    #pragma unroll
    for (int c = 0; c < 64; ++c) acc += r[c] * Wc[c];
    out[n] = acc;
}

// ---------------------------------------------------------------------------
// Host orchestration
// ---------------------------------------------------------------------------
struct LayerW {
    const float *Wk, *bk, *Wq, *bq, *Wv, *bv, *a_rel, *m_rel, *p_rel, *Wa, *ba;
    int din;
};

extern "C" void kernel_launch(void* const* d_in, const int* in_sizes, int n_in,
                              void* d_out, int out_size, void* d_ws, size_t ws_size,
                              hipStream_t stream) {
    (void)n_in; (void)out_size; (void)ws_size;
    const float* x_tx   = (const float*)d_in[0];
    const float* x_m    = (const float*)d_in[1];
    const int*   tm_src = (const int*)d_in[2];
    const int*   tm_dst = (const int*)d_in[3];
    const int*   mt_src = (const int*)d_in[4];
    const int*   mt_dst = (const int*)d_in[5];
    const int N_TX = in_sizes[0] / 128;
    const int N_M  = in_sizes[1] / 128;
    const int E    = in_sizes[2];

    LayerW L1 = {(const float*)d_in[6],  (const float*)d_in[7],  (const float*)d_in[8],
                 (const float*)d_in[9],  (const float*)d_in[10], (const float*)d_in[11],
                 (const float*)d_in[12], (const float*)d_in[13], (const float*)d_in[14],
                 (const float*)d_in[15], (const float*)d_in[16], 128};
    LayerW L2 = {(const float*)d_in[17], (const float*)d_in[18], (const float*)d_in[19],
                 (const float*)d_in[20], (const float*)d_in[21], (const float*)d_in[22],
                 (const float*)d_in[23], (const float*)d_in[24], (const float*)d_in[25],
                 (const float*)d_in[26], (const float*)d_in[27], 64};
    const float* skip2 = (const float*)d_in[28];
    const float* Wc    = (const float*)d_in[29];
    const float* bc    = (const float*)d_in[30];

    // ---- workspace layout -------------------------------------------------
    char* ws = (char*)d_ws;
    size_t off = 0;
    auto alloc = [&](size_t bytes) -> char* {
        char* p = ws + off;
        off += (bytes + 255) & ~(size_t)255;
        return p;
    };
    unsigned short* wqT  = (unsigned short*)alloc((size_t)2 * 64 * 128 * 2);
    unsigned short* wkpT = (unsigned short*)alloc((size_t)2 * 64 * 128 * 2);
    unsigned short* wvpT = (unsigned short*)alloc((size_t)2 * 64 * 128 * 2);
    unsigned short* waT  = (unsigned short*)alloc((size_t)2 * 64 * 64 * 2);
    float* bkp   = (float*)alloc((size_t)2 * 64 * 4);
    float* bvp   = (float*)alloc((size_t)2 * 64 * 4);
    unsigned short* xb_tx = (unsigned short*)alloc((size_t)N_TX * 128 * 2);  // bf16 acts
    unsigned short* xb_m  = (unsigned short*)alloc((size_t)N_M * 128 * 2);
    unsigned short* ab_tx = (unsigned short*)alloc((size_t)N_TX * 64 * 2);   // bf16 gelu
    unsigned short* ab_m  = (unsigned short*)alloc((size_t)N_M * 64 * 2);
    float* Q_tx  = (float*)alloc((size_t)N_TX * 64 * 4);
    float* Kp_tx = (float*)alloc((size_t)N_TX * 64 * 4);
    float* Vp_tx = (float*)alloc((size_t)N_TX * 64 * 4);
    float* Q_m   = (float*)alloc((size_t)N_M * 64 * 4);
    float* Kp_m  = (float*)alloc((size_t)N_M * 64 * 4);
    float* Vp_m  = (float*)alloc((size_t)N_M * 64 * 4);
    float* h1_tx = (float*)alloc((size_t)N_TX * 64 * 4);
    float* h1_m  = (float*)alloc((size_t)N_M * 64 * 4);
    float* agg_tx = (float*)alloc((size_t)N_TX * 64 * 4);
    float* agg_m  = (float*)alloc((size_t)N_M * 64 * 4);
    unsigned* amax_tx = (unsigned*)alloc((size_t)N_TX * 4 * 4);
    float*    den_tx  = (float*)alloc((size_t)N_TX * 4 * 4);
    unsigned* amax_m  = (unsigned*)alloc((size_t)N_M * 4 * 4);
    float*    den_m   = (float*)alloc((size_t)N_M * 4 * 4);
    float* alphabuf = (float*)alloc((size_t)E * 4 * 4);
    float* h2_tx = Q_tx;  // Q dead by the time layer-2 outputs are written
    float* h2_m  = Q_m;

    auto zero = [&](void* p, size_t n_u32) {
        fill_u32<<<dim3((unsigned)((n_u32 + 255) / 256)), dim3(256), 0, stream>>>(
            (unsigned*)p, 0u, n_u32);
    };
    auto cvt = [&](const float* in, unsigned short* out, size_t n) {
        cvt_bf16<<<dim3((unsigned)((n + 255) / 256)), dim3(256), 0, stream>>>(in, out, n);
    };
    auto gemm = [&](const unsigned short* A, const unsigned short* Bt, const float* bias,
                    float* C, int N, int K, int mode, const float* skip, const float* xin) {
        gemm_wmma<<<dim3((N + 63) / 64), dim3(128), 0, stream>>>(
            A, Bt, bias, C, N, K, mode, skip, xin);
    };
    const int eb = (E * 4 + 255) / 256;

    auto run_layer = [&](const LayerW& L, const float* xin_tx, const float* xin_m,
                         float* out_tx, float* out_m, int mode) {
        const int din = L.din;
        cvt(xin_tx, xb_tx, (size_t)N_TX * din);
        cvt(xin_m,  xb_m,  (size_t)N_M * din);
        prep_qkv<<<dim3(2), dim3(256), 0, stream>>>(L.Wk, L.bk, L.Wq, L.bq, L.Wv, L.bv,
                                                    L.a_rel, L.m_rel, din,
                                                    wqT, wkpT, wvpT, bkp, bvp);
        prep_wa<<<dim3(2), dim3(256), 0, stream>>>(L.Wa, waT);

        // projections (fused relation transforms already folded into K/V weights)
        gemm(xb_tx, wqT,             L.bq,      Q_tx,  N_TX, din, 0, nullptr, nullptr);
        gemm(xb_tx, wkpT,            bkp,       Kp_tx, N_TX, din, 0, nullptr, nullptr);
        gemm(xb_tx, wvpT,            bvp,       Vp_tx, N_TX, din, 0, nullptr, nullptr);
        gemm(xb_m,  wqT  + 64 * din, L.bq + 64, Q_m,   N_M,  din, 0, nullptr, nullptr);
        gemm(xb_m,  wkpT + 64 * din, bkp + 64,  Kp_m,  N_M,  din, 0, nullptr, nullptr);
        gemm(xb_m,  wvpT + 64 * din, bvp + 64,  Vp_m,  N_M,  din, 0, nullptr, nullptr);

        // relation 0: tx -> m
        zero(amax_m, (size_t)N_M * 4);
        zero(den_m,  (size_t)N_M * 4);
        zero(agg_m,  (size_t)N_M * 64);
        edge_alpha<<<dim3(eb), dim3(256), 0, stream>>>(Q_m, Kp_tx, tm_src, tm_dst, E,
                                                       L.p_rel, alphabuf, amax_m);
        edge_accum<<<dim3(eb), dim3(256), 0, stream>>>(Vp_tx, tm_src, tm_dst, E,
                                                       alphabuf, amax_m, den_m, agg_m);
        // relation 1: m -> tx
        zero(amax_tx, (size_t)N_TX * 4);
        zero(den_tx,  (size_t)N_TX * 4);
        zero(agg_tx,  (size_t)N_TX * 64);
        edge_alpha<<<dim3(eb), dim3(256), 0, stream>>>(Q_tx, Kp_m, mt_src, mt_dst, E,
                                                       L.p_rel + 4, alphabuf, amax_tx);
        edge_accum<<<dim3(eb), dim3(256), 0, stream>>>(Vp_m, mt_src, mt_dst, E,
                                                       alphabuf, amax_tx, den_tx, agg_tx);

        node_finish<<<dim3((N_TX * 64 + 255) / 256), dim3(256), 0, stream>>>(
            agg_tx, den_tx, ab_tx, N_TX);
        node_finish<<<dim3((N_M * 64 + 255) / 256), dim3(256), 0, stream>>>(
            agg_m, den_m, ab_m, N_M);

        // output projection + per-layer epilogue (relu / sigmoid skip)
        gemm(ab_tx, waT,           L.ba,      out_tx, N_TX, 64, mode, skip2,     xin_tx);
        gemm(ab_m,  waT + 64 * 64, L.ba + 64, out_m,  N_M,  64, mode, skip2 + 1, xin_m);
    };

    run_layer(L1, x_tx, x_m, h1_tx, h1_m, /*relu*/ 1);
    run_layer(L2, h1_tx, h1_m, h2_tx, h2_m, /*skip*/ 2);

    classify<<<dim3((N_TX + 255) / 256), dim3(256), 0, stream>>>(h2_tx, Wc, bc,
                                                                 (float*)d_out, N_TX);
}
